// MultiHeadAttention_46420006535896
// MI455X (gfx1250) — matrix operand order repaired and checked by reading, not compiled, not run on hardware
//
#include <hip/hip_runtime.h>
#include <hip/hip_bf16.h>

// MI455X / gfx1250: wave32, WMMA path (v_wmma_f32_16x16x32_bf16).
// MHA forward: B=2, S=2048, D=1024, H=16, DH=64.
// Outputs (concat in d_out): out [B,S,D] f32, attn [B,H,S,S] f32.
//
// Pipeline:
//   cvt:    q_in/k_in/v_in f32 -> bf16 (elementwise)
//   wt:     Wq/Wk/Wv/Wo f32 [K][N] -> bf16 transposed [N][K]
//   proj:   Q/K -> [B,H,S,DH] bf16; V -> [B,H,DH,S] bf16 (K-contig for ctx)
//   logits: QK^T/8 + mask*-1e9 -> attn f32 (d_out)
//   softmax row-wise in place
//   ctx:    attn @ V -> [B,S,D] bf16
//   out:    ctx @ Wo + bo -> f32 (d_out)
// All GEMM fragments are contiguous b128 loads per the ISA WMMA VGPR layouts.

typedef __attribute__((ext_vector_type(16))) __bf16       v16bf;
typedef __attribute__((ext_vector_type(8)))  float        v8f;
typedef __attribute__((ext_vector_type(4)))  unsigned int u32x4;

#define B_     2
#define S_     2048
#define D_     1024
#define H_     16
#define DH_    64
#define MROWS_ (B_ * S_)   // 4096

union FragBF {
  v16bf bf;
  u32x4 q[2];
  unsigned int u[8];
};

static __device__ __forceinline__ unsigned int pack_bf(float lo, float hi) {
  unsigned int a = __builtin_bit_cast(unsigned int, lo);
  unsigned int b = __builtin_bit_cast(unsigned int, hi);
  a += 0x7FFFu + ((a >> 16) & 1u);
  b += 0x7FFFu + ((b >> 16) & 1u);
  return (a >> 16) | (b & 0xFFFF0000u);
}
static __device__ __forceinline__ __bf16 to_bf(float f) {
  unsigned int u = __builtin_bit_cast(unsigned int, f);
  u += 0x7FFFu + ((u >> 16) & 1u);
  return __builtin_bit_cast(__bf16, (unsigned short)(u >> 16));
}

static __device__ __forceinline__ v8f wmma_bf16(v16bf a, v16bf b, v8f c) {
  return __builtin_amdgcn_wmma_f32_16x16x32_bf16(false, a, false, b,
                                                 (short)0, c, false, false);
}

// A 16x32 (MxK): lane row = lane&15, khalf = lane>>4; two contiguous 8-elt
//   chunks at K = 8*khalf and 16+8*khalf.
// B 32x16 (KxN): lane col = lane&15; two contiguous 8-elt chunks at K = 8*khalf and 16+8*khalf.
// C/D 16x16 f32: VGPR r -> M = r + 8*(lane>>4), N = lane&15.
static __device__ __forceinline__ v16bf load_a_bf16(const __bf16* ap, int khalf) {
  FragBF f;
  f.q[0] = *(const u32x4*)(ap + 8 * khalf);
  f.q[1] = *(const u32x4*)(ap + 16 + 8 * khalf);
  return f.bf;
}
static __device__ __forceinline__ v16bf load_a_f32(const float* ap, int khalf) {
  FragBF f;
#pragma unroll
  for (int i = 0; i < 8; ++i) {
    int k = ((2 * i) < 8 ? 2 * i : 2 * i + 8) + 8 * khalf;
    f.u[i] = pack_bf(ap[k], ap[k + 1]);
  }
  return f.bf;
}
static __device__ __forceinline__ v16bf load_b_bf16(const __bf16* bp, int khalf) {
  FragBF f;
  f.q[0] = *(const u32x4*)(bp + 8 * khalf);
  f.q[1] = *(const u32x4*)(bp + 16 + 8 * khalf);
  return f.bf;
}

// ---------------------------------------------------------------------------
// Elementwise f32 -> bf16 (float4 in, uint2 out)
// ---------------------------------------------------------------------------
__global__ __launch_bounds__(256) void cvt_kernel(const float* __restrict__ in,
                                                  __bf16* __restrict__ out, int n4) {
  int i = blockIdx.x * 256 + threadIdx.x;
  if (i < n4) {
    float4 v = ((const float4*)in)[i];
    uint2 p;
    p.x = pack_bf(v.x, v.y);
    p.y = pack_bf(v.z, v.w);
    ((uint2*)out)[i] = p;
  }
}

// ---------------------------------------------------------------------------
// Weight transpose+convert: W f32 [K=D][N=D] -> Wt bf16 [N][K]
// 32x32 LDS tile; 256 threads.
// ---------------------------------------------------------------------------
__global__ __launch_bounds__(256) void wt_kernel(const float* __restrict__ W,
                                                 __bf16* __restrict__ Wt) {
  __shared__ __bf16 tile[32][33];
  const int n0 = blockIdx.x * 32;
  const int k0 = blockIdx.y * 32;
  const int tx = threadIdx.x & 31;
  const int ty = threadIdx.x >> 5;   // 0..7
#pragma unroll
  for (int r = 0; r < 4; ++r) {
    int k = ty + r * 8;
    tile[k][tx] = to_bf(W[(size_t)(k0 + k) * D_ + n0 + tx]);
  }
  __syncthreads();
#pragma unroll
  for (int r = 0; r < 4; ++r) {
    int n = ty + r * 8;
    Wt[(size_t)(n0 + n) * D_ + k0 + tx] = tile[tx][n];
  }
}

// ---------------------------------------------------------------------------
// Kernel: QKV projection. out = X @ Wt^T + bias (bf16 in, bf16 out).
// VT=0: store [B,H,S,DH] (Q,K).  VT=1: store [B,H,DH,S] (V).
// 8 waves (2M x 4N), wave tile 64x64, block tile 128x256.
// ---------------------------------------------------------------------------
template <int VT>
__global__ __launch_bounds__(256) void proj_qkv_kernel(
    const __bf16* __restrict__ X, const __bf16* __restrict__ Wt,
    const float* __restrict__ bias, __bf16* __restrict__ outh) {
  const int lane  = threadIdx.x & 31;
  const int wave  = threadIdx.x >> 5;
  const int row   = lane & 15;
  const int khalf = lane >> 4;
  const int m0 = blockIdx.x * 128 + (wave >> 2) * 64;
  const int n0 = blockIdx.y * 256 + (wave & 3) * 64;

  v8f acc[4][4] = {};

  for (int k0 = 0; k0 < D_; k0 += 32) {
    v16bf a[4], b[4];
#pragma unroll
    for (int i = 0; i < 4; ++i)
      a[i] = load_a_bf16(X + (size_t)(m0 + 16 * i + row) * D_ + k0, khalf);
#pragma unroll
    for (int j = 0; j < 4; ++j)
      b[j] = load_b_bf16(Wt + (size_t)(n0 + 16 * j + row) * D_ + k0, khalf);
#pragma unroll
    for (int i = 0; i < 4; ++i)
#pragma unroll
      for (int j = 0; j < 4; ++j)
        acc[i][j] = wmma_bf16(a[i], b[j], acc[i][j]);
  }

  float bv[4];
#pragma unroll
  for (int j = 0; j < 4; ++j) bv[j] = bias[n0 + 16 * j + row];

#pragma unroll
  for (int i = 0; i < 4; ++i)
#pragma unroll
    for (int j = 0; j < 4; ++j)
#pragma unroll
      for (int r = 0; r < 8; ++r) {
        int gm = m0 + 16 * i + r + 8 * khalf;     // row in [B*S]
        int gn = n0 + 16 * j + row;               // col in [D]
        float v = acc[i][j][r] + bv[j];
        int bb = gm >> 11, ss = gm & (S_ - 1);
        int hh = gn >> 6,  dd = gn & (DH_ - 1);
        size_t idx = VT
          ? (((size_t)(bb * H_ + hh)) * DH_ + dd) * S_ + ss    // [B,H,DH,S]
          : (((size_t)(bb * H_ + hh)) * S_ + ss) * DH_ + dd;   // [B,H,S,DH]
        outh[idx] = to_bf(v);
      }
}

// ---------------------------------------------------------------------------
// Kernel: logits = (Q Kh^T) / sqrt(DH) + mask * -1e9 -> attn (f32)
// Grid: (S/128, S/256, B*H). Wave tile 64x64; K loop = DH = 64 (2 steps).
// ---------------------------------------------------------------------------
__global__ __launch_bounds__(256) void logits_kernel(
    const __bf16* __restrict__ Qh, const __bf16* __restrict__ Kh,
    const float* __restrict__ mask, float* __restrict__ attn) {
  const int lane  = threadIdx.x & 31;
  const int wave  = threadIdx.x >> 5;
  const int row   = lane & 15;
  const int khalf = lane >> 4;
  const int bh = blockIdx.z;
  const int bb = bh >> 4;  // H_ == 16
  const __bf16* q  = Qh + (size_t)bh * S_ * DH_;
  const __bf16* kk = Kh + (size_t)bh * S_ * DH_;
  float* att       = attn + (size_t)bh * S_ * S_;
  const float* msk = mask + (size_t)bb * S_ * S_;

  const int m0 = blockIdx.x * 128 + (wave >> 2) * 64;   // q index
  const int n0 = blockIdx.y * 256 + (wave & 3) * 64;    // k index

  v8f acc[4][4] = {};
#pragma unroll
  for (int k0 = 0; k0 < DH_; k0 += 32) {
    v16bf a[4], b[4];
#pragma unroll
    for (int i = 0; i < 4; ++i)
      a[i] = load_a_bf16(q + (size_t)(m0 + 16 * i + row) * DH_ + k0, khalf);
#pragma unroll
    for (int j = 0; j < 4; ++j)   // B = K^T: rows of Kh are contiguous in K
      b[j] = load_b_bf16(kk + (size_t)(n0 + 16 * j + row) * DH_ + k0, khalf);
#pragma unroll
    for (int i = 0; i < 4; ++i)
#pragma unroll
      for (int j = 0; j < 4; ++j)
        acc[i][j] = wmma_bf16(a[i], b[j], acc[i][j]);
  }

  const float scale = 0.125f;  // 1/sqrt(64)
#pragma unroll
  for (int i = 0; i < 4; ++i)
#pragma unroll
    for (int j = 0; j < 4; ++j)
#pragma unroll
      for (int r = 0; r < 8; ++r) {
        int gm = m0 + 16 * i + r + 8 * khalf;
        int gn = n0 + 16 * j + row;
        att[(size_t)gm * S_ + gn] =
            acc[i][j][r] * scale + msk[(size_t)gm * S_ + gn] * -1e9f;
      }
}

// ---------------------------------------------------------------------------
// Kernel: row softmax in place. One 256-thread block per row (len 2048).
// ---------------------------------------------------------------------------
__global__ __launch_bounds__(256) void softmax_kernel(float* __restrict__ attn) {
  const int tid  = threadIdx.x;
  const int lane = tid & 31;
  const int wave = tid >> 5;
  float* rowp = attn + (size_t)blockIdx.x * S_;
  __shared__ float red[8];

  float4 va = ((const float4*)rowp)[tid * 2 + 0];
  float4 vb = ((const float4*)rowp)[tid * 2 + 1];

  float mx = fmaxf(fmaxf(fmaxf(va.x, va.y), fmaxf(va.z, va.w)),
                   fmaxf(fmaxf(vb.x, vb.y), fmaxf(vb.z, vb.w)));
#pragma unroll
  for (int off = 16; off > 0; off >>= 1)
    mx = fmaxf(mx, __shfl_xor(mx, off, 32));
  if (lane == 0) red[wave] = mx;
  __syncthreads();
  float rmax = red[0];
#pragma unroll
  for (int w = 1; w < 8; ++w) rmax = fmaxf(rmax, red[w]);
  __syncthreads();

  va.x = __expf(va.x - rmax); va.y = __expf(va.y - rmax);
  va.z = __expf(va.z - rmax); va.w = __expf(va.w - rmax);
  vb.x = __expf(vb.x - rmax); vb.y = __expf(vb.y - rmax);
  vb.z = __expf(vb.z - rmax); vb.w = __expf(vb.w - rmax);
  float sum = (va.x + va.y + va.z + va.w) + (vb.x + vb.y + vb.z + vb.w);
#pragma unroll
  for (int off = 16; off > 0; off >>= 1)
    sum += __shfl_xor(sum, off, 32);
  if (lane == 0) red[wave] = sum;
  __syncthreads();
  float tot = 0.f;
#pragma unroll
  for (int w = 0; w < 8; ++w) tot += red[w];
  float inv = 1.0f / tot;

  va.x *= inv; va.y *= inv; va.z *= inv; va.w *= inv;
  vb.x *= inv; vb.y *= inv; vb.z *= inv; vb.w *= inv;
  ((float4*)rowp)[tid * 2 + 0] = va;
  ((float4*)rowp)[tid * 2 + 1] = vb;
}

// ---------------------------------------------------------------------------
// Kernel: ctx = attn @ V per head -> ctx_concat [B,S,D] (bf16)
// V stored transposed [B,H,DH,S] => B fragments are K-contiguous b128 loads.
// Grid: (S/512, B*H). 8 waves in M; wave tile 64x64 (full DH).
// ---------------------------------------------------------------------------
__global__ __launch_bounds__(256) void ctx_kernel(
    const float* __restrict__ attn, const __bf16* __restrict__ Vt,
    __bf16* __restrict__ ctx) {
  const int lane  = threadIdx.x & 31;
  const int wave  = threadIdx.x >> 5;
  const int row   = lane & 15;
  const int khalf = lane >> 4;
  const int bh = blockIdx.y;
  const int bb = bh >> 4, hh = bh & 15;
  const float*  att = attn + (size_t)bh * S_ * S_;
  const __bf16* vt  = Vt + (size_t)bh * DH_ * S_;   // [DH][S]

  const int m0 = blockIdx.x * 512 + wave * 64;  // q index

  v8f acc[4][4] = {};
  for (int k0 = 0; k0 < S_; k0 += 32) {
    v16bf a[4], b[4];
#pragma unroll
    for (int i = 0; i < 4; ++i)
      a[i] = load_a_f32(att + (size_t)(m0 + 16 * i + row) * S_ + k0, khalf);
#pragma unroll
    for (int j = 0; j < 4; ++j)   // B[kk][n] = Vt[n][k0+kk], contiguous in kk
      b[j] = load_b_bf16(vt + (size_t)(16 * j + row) * S_ + k0, khalf);
#pragma unroll
    for (int i = 0; i < 4; ++i)
#pragma unroll
      for (int j = 0; j < 4; ++j)
        acc[i][j] = wmma_bf16(a[i], b[j], acc[i][j]);
  }

#pragma unroll
  for (int i = 0; i < 4; ++i)
#pragma unroll
    for (int j = 0; j < 4; ++j)
#pragma unroll
      for (int r = 0; r < 8; ++r) {
        int gq = m0 + 16 * i + r + 8 * khalf;   // q position
        int gn = 16 * j + row;                  // dh
        ctx[((size_t)(bb * S_ + gq)) * D_ + hh * DH_ + gn] = to_bf(acc[i][j][r]);
      }
}

// ---------------------------------------------------------------------------
// Kernel: out = ctx @ Wo + bo -> f32 row-major [B*S, D]
// ---------------------------------------------------------------------------
__global__ __launch_bounds__(256) void out_proj_kernel(
    const __bf16* __restrict__ Xb, const __bf16* __restrict__ Wt,
    const float* __restrict__ bias, float* __restrict__ out) {
  const int lane  = threadIdx.x & 31;
  const int wave  = threadIdx.x >> 5;
  const int row   = lane & 15;
  const int khalf = lane >> 4;
  const int m0 = blockIdx.x * 128 + (wave >> 2) * 64;
  const int n0 = blockIdx.y * 256 + (wave & 3) * 64;

  v8f acc[4][4] = {};
  for (int k0 = 0; k0 < D_; k0 += 32) {
    v16bf a[4], b[4];
#pragma unroll
    for (int i = 0; i < 4; ++i)
      a[i] = load_a_bf16(Xb + (size_t)(m0 + 16 * i + row) * D_ + k0, khalf);
#pragma unroll
    for (int j = 0; j < 4; ++j)
      b[j] = load_b_bf16(Wt + (size_t)(n0 + 16 * j + row) * D_ + k0, khalf);
#pragma unroll
    for (int i = 0; i < 4; ++i)
#pragma unroll
      for (int j = 0; j < 4; ++j)
        acc[i][j] = wmma_bf16(a[i], b[j], acc[i][j]);
  }

  float bv[4];
#pragma unroll
  for (int j = 0; j < 4; ++j) bv[j] = bias[n0 + 16 * j + row];

#pragma unroll
  for (int i = 0; i < 4; ++i)
#pragma unroll
    for (int j = 0; j < 4; ++j)
#pragma unroll
      for (int r = 0; r < 8; ++r) {
        int gm = m0 + 16 * i + r + 8 * khalf;
        int gn = n0 + 16 * j + row;
        out[(size_t)gm * D_ + gn] = acc[i][j][r] + bv[j];
      }
}

// ---------------------------------------------------------------------------
extern "C" void kernel_launch(void* const* d_in, const int* in_sizes, int n_in,
                              void* d_out, int out_size, void* d_ws, size_t ws_size,
                              hipStream_t stream) {
  (void)in_sizes; (void)n_in; (void)out_size; (void)ws_size;
  const float* q_in = (const float*)d_in[0];
  const float* k_in = (const float*)d_in[1];
  const float* v_in = (const float*)d_in[2];
  const float* mask = (const float*)d_in[3];
  const float* Wq   = (const float*)d_in[4];
  const float* bq   = (const float*)d_in[5];
  const float* Wk   = (const float*)d_in[6];
  const float* bk   = (const float*)d_in[7];
  const float* Wv   = (const float*)d_in[8];
  const float* bv   = (const float*)d_in[9];
  const float* Wo   = (const float*)d_in[10];
  const float* bo   = (const float*)d_in[11];

  const size_t pe = (size_t)B_ * S_ * D_;    // 4M elements
  const size_t we = (size_t)D_ * D_;         // 1M elements
  __bf16* xq  = (__bf16*)d_ws;               // [B*S, D] bf16
  __bf16* xk  = xq + pe;
  __bf16* xv  = xk + pe;
  __bf16* qh  = xv + pe;                     // [B,H,S,DH]
  __bf16* kh  = qh + pe;                     // [B,H,S,DH]
  __bf16* vt  = kh + pe;                     // [B,H,DH,S]
  __bf16* ctx = vt + pe;                     // [B,S,D]
  __bf16* wqt = ctx + pe;                    // [N][K] bf16
  __bf16* wkt = wqt + we;
  __bf16* wvt = wkt + we;
  __bf16* wot = wvt + we;

  float* out  = (float*)d_out;               // [B,S,D] f32
  float* attn = out + pe;                    // [B,H,S,S] f32

  dim3 blk(256);

  const int n4 = (int)(pe / 4);
  cvt_kernel<<<dim3((n4 + 255) / 256), blk, 0, stream>>>(q_in, xq, n4);
  cvt_kernel<<<dim3((n4 + 255) / 256), blk, 0, stream>>>(k_in, xk, n4);
  cvt_kernel<<<dim3((n4 + 255) / 256), blk, 0, stream>>>(v_in, xv, n4);

  dim3 gwt(D_ / 32, D_ / 32);                // 32 x 32
  wt_kernel<<<gwt, blk, 0, stream>>>(Wq, wqt);
  wt_kernel<<<gwt, blk, 0, stream>>>(Wk, wkt);
  wt_kernel<<<gwt, blk, 0, stream>>>(Wv, wvt);
  wt_kernel<<<gwt, blk, 0, stream>>>(Wo, wot);

  dim3 gproj(MROWS_ / 128, D_ / 256);        // 32 x 4
  proj_qkv_kernel<0><<<gproj, blk, 0, stream>>>(xq, wqt, bq, qh);
  proj_qkv_kernel<0><<<gproj, blk, 0, stream>>>(xk, wkt, bk, kh);
  proj_qkv_kernel<1><<<gproj, blk, 0, stream>>>(xv, wvt, bv, vt);

  dim3 glog(S_ / 128, S_ / 256, B_ * H_);    // 16 x 8 x 32
  logits_kernel<<<glog, blk, 0, stream>>>(qh, kh, mask, attn);

  softmax_kernel<<<dim3(B_ * H_ * S_), blk, 0, stream>>>(attn);

  dim3 gctx(S_ / 512, B_ * H_);              // 4 x 32
  ctx_kernel<<<gctx, blk, 0, stream>>>(attn, vt, ctx);

  out_proj_kernel<<<gproj, blk, 0, stream>>>(ctx, wot, bo, out);
}
